// TSDeformConv2d_76347338654200
// MI455X (gfx1250) — matrix operand order repaired and checked by reading, not compiled, not run on hardware
//
#include <hip/hip_runtime.h>

#define BATCH 8
#define H 192
#define W 96
#define S 7
#define HID 64
#define MPOS (BATCH * H * W)   // 147456 positions total
#define MIMG (H * W)           // 18432 positions per image

typedef __attribute__((ext_vector_type(16))) _Float16 v16h;
typedef __attribute__((ext_vector_type(8)))  _Float16 v8h;
typedef __attribute__((ext_vector_type(8)))  float    v8f;

// ---------------------------------------------------------------------------
// BN folding: h = (acc + b1 - mean) * rsqrt(var+eps) * gamma + beta
//           = acc * scale + shift
__global__ void bn_prep_kernel(const float* __restrict__ gamma,
                               const float* __restrict__ beta,
                               const float* __restrict__ mean,
                               const float* __restrict__ var,
                               const float* __restrict__ b1,
                               float* __restrict__ scale,
                               float* __restrict__ shift) {
    int n = threadIdx.x;
    if (n < HID) {
        float s = gamma[n] * rsqrtf(var[n] + 1e-5f);
        scale[n] = s;
        shift[n] = (b1[n] - mean[n]) * s + beta[n];
    }
}

// ---------------------------------------------------------------------------
// Pack OIHW f32 weights into per-lane WMMA B fragments (f16).
// K ordering: k = tap*C + c  (tap = ky*3+kx), matching the col builders.
// B fragment layout: lane l = column n = ntile*16 + (l&15);
//   elems 0..15 = K = kstep*32 + ((l>=16)?16:0) + e
__global__ void pack_weight_kernel(const float* __restrict__ w, v16h* __restrict__ dst,
                                   int C, int Nreal, int Kreal, int ntiles, int ksteps) {
    int idx = blockIdx.x * blockDim.x + threadIdx.x;
    int total = ntiles * ksteps * 32;
    if (idx >= total) return;
    int lane = idx & 31;
    int ks   = (idx >> 5) % ksteps;
    int nt   = (idx >> 5) / ksteps;
    int n  = nt * 16 + (lane & 15);
    int kb = ks * 32 + ((lane >> 4) << 4);
    v16h frag;
#pragma unroll
    for (int e = 0; e < 16; ++e) {
        int k = kb + e;
        float v = 0.0f;
        if (k < Kreal && n < Nreal) {
            int t = k / C, c = k % C;               // k = t*C + c
            v = w[(size_t)n * (C * 9) + c * 9 + t]; // OIHW: [n][c][ky][kx]
        }
        frag[e] = (_Float16)v;
    }
    dst[(size_t)(nt * ksteps + ks) * 32 + lane] = frag;
}

// ---------------------------------------------------------------------------
// off1 = conv3x3(x, w_off1) + b_off1  (tiny: 7 in, 18 out) -> [pos][18] f32
__global__ void off1_conv_kernel(const float* __restrict__ x,
                                 const float* __restrict__ w,
                                 const float* __restrict__ bias,
                                 float* __restrict__ off1) {
    int idx = blockIdx.x * blockDim.x + threadIdx.x;
    if (idx >= MPOS * 18) return;
    int oc  = idx % 18;
    int pos = idx / 18;
    int xw = pos % W; int tt = pos / W; int yh = tt % H; int b = tt / H;
    const float* xb = x + (size_t)b * MIMG * S;
    const float* pw = w + (size_t)oc * (S * 9);
    float acc = bias[oc];
    for (int ky = 0; ky < 3; ++ky) {
        int yy = yh + ky - 1;
        if (yy < 0 || yy >= H) continue;
        for (int kx = 0; kx < 3; ++kx) {
            int xx = xw + kx - 1;
            if (xx < 0 || xx >= W) continue;
            const float* px = xb + ((size_t)yy * W + xx) * S;
            int tap = ky * 3 + kx;
#pragma unroll
            for (int c = 0; c < S; ++c)
                acc += px[c] * pw[c * 9 + tap];
        }
    }
    off1[(size_t)pos * 18 + oc] = acc;
}

// ---------------------------------------------------------------------------
// Deformable im2col for stage 1: x (NHWC f32, C=7) -> col1[pos][64] f16
// k index = tap*7 + c ; element 63 zero-padded.
__global__ void col1_build_kernel(const float* __restrict__ x,
                                  const float* __restrict__ off1,
                                  _Float16* __restrict__ col1) {
    int idx = blockIdx.x * blockDim.x + threadIdx.x;
    if (idx >= MPOS * 9) return;
    int t = idx % 9; int pos = idx / 9;
    int xw = pos % W; int tmp = pos / W; int yh = tmp % H; int b = tmp / H;
    float dy = off1[(size_t)pos * 18 + t * 2];
    float dx = off1[(size_t)pos * 18 + t * 2 + 1];
    float py = (float)(yh + t / 3 - 1) + dy;
    float px = (float)(xw + t % 3 - 1) + dx;
    float fy = floorf(py), fx = floorf(px);
    int y0 = (int)fy, x0 = (int)fx;
    float ly = py - fy, lx = px - fx;
    float w00 = (1.f - ly) * (1.f - lx), w01 = (1.f - ly) * lx;
    float w10 = ly * (1.f - lx),         w11 = ly * lx;
    const float* xb = x + (size_t)b * MIMG * S;
    float s[S];
#pragma unroll
    for (int c = 0; c < S; ++c) s[c] = 0.f;
    auto corner = [&](int yy, int xx, float wgt) {
        if (yy >= 0 && yy < H && xx >= 0 && xx < W) {
            const float* p = xb + ((size_t)yy * W + xx) * S;
#pragma unroll
            for (int c = 0; c < S; ++c) s[c] += wgt * p[c];
        }
    };
    corner(y0,     x0,     w00);
    corner(y0,     x0 + 1, w01);
    corner(y0 + 1, x0,     w10);
    corner(y0 + 1, x0 + 1, w11);
    _Float16* dst = col1 + (size_t)pos * 64 + t * S;
#pragma unroll
    for (int c = 0; c < S; ++c) dst[c] = (_Float16)s[c];
    if (t == 0) col1[(size_t)pos * 64 + 63] = (_Float16)0.0f;
}

// ---------------------------------------------------------------------------
// WMMA GEMM core: one wave computes a 16(M) x (NT*16)(N) tile over K = KS*32.
// A: row-major f16 [M][KS*32]; Bp: prepacked fragments [NT][KS][32 lanes].
template <int NT, int KS>
__device__ __forceinline__ void gemm_core(const _Float16* __restrict__ A,
                                          const v16h* __restrict__ Bp,
                                          int mtile, int lane, v8f (&acc)[NT]) {
    const int row = lane & 15;
    const int kba = (lane >> 4) << 4;  // 0 or 16
    const _Float16* arow = A + (size_t)(mtile * 16 + row) * (KS * 32);
    for (int ks = 0; ks < KS; ++ks) {
        v8h a0 = *(const v8h*)(arow + ks * 32 + kba);       // K = kb .. kb+7
        v8h a1 = *(const v8h*)(arow + ks * 32 + kba + 8);   // K = kb+8 .. kb+15
        v16h a = __builtin_shufflevector(a0, a1, 0,1,2,3,4,5,6,7,8,9,10,11,12,13,14,15);
#pragma unroll
        for (int nt = 0; nt < NT; ++nt) {
            v16h bfrag = Bp[((size_t)nt * KS + ks) * 32 + lane];
            acc[nt] = __builtin_amdgcn_wmma_f32_16x16x32_f16(
                false, a, false, bfrag, (short)0, acc[nt], false, false);
        }
    }
}

// GEMM1: col1[M,64] x w1[64,64] with fused BN+ReLU -> h (f16, NHWC C=64)
__global__ void gemm1_kernel(const _Float16* __restrict__ A, const v16h* __restrict__ Bp,
                             const float* __restrict__ scale, const float* __restrict__ shift,
                             _Float16* __restrict__ hout) {
    int wave = (int)((blockIdx.x * blockDim.x + threadIdx.x) >> 5);
    int lane = threadIdx.x & 31;
    v8f acc[4];
#pragma unroll
    for (int nt = 0; nt < 4; ++nt) acc[nt] = (v8f){0,0,0,0,0,0,0,0};
    gemm_core<4, 2>(A, Bp, wave, lane, acc);
    int n0 = lane & 15;
    int mh = (lane >> 4) << 3;
#pragma unroll
    for (int nt = 0; nt < 4; ++nt) {
        int n = nt * 16 + n0;
        float sc = scale[n], sh = shift[n];
#pragma unroll
        for (int r = 0; r < 8; ++r) {
            int m = wave * 16 + mh + r;
            float v = acc[nt][r] * sc + sh;
            hout[(size_t)m * HID + n] = (_Float16)(v > 0.f ? v : 0.f);
        }
    }
}

// ---------------------------------------------------------------------------
// Standard 3x3 im2col of h (one image) -> col[pos][576] f16, k = tap*64 + c
__global__ void colstd_build_kernel(const _Float16* __restrict__ h, int b,
                                    _Float16* __restrict__ col) {
    int idx = blockIdx.x * blockDim.x + threadIdx.x;
    if (idx >= MIMG * 9) return;
    int t = idx % 9; int pos = idx / 9;
    int xw = pos % W; int yh = pos / W;
    int yy = yh + t / 3 - 1, xx = xw + t % 3 - 1;
    uint4* d4 = (uint4*)(col + (size_t)pos * 576 + t * HID);
    if (yy >= 0 && yy < H && xx >= 0 && xx < W) {
        const uint4* s4 = (const uint4*)(h + (((size_t)b * H + yy) * W + xx) * HID);
#pragma unroll
        for (int i = 0; i < 8; ++i) d4[i] = s4[i];
    } else {
        uint4 z = {0u, 0u, 0u, 0u};
#pragma unroll
        for (int i = 0; i < 8; ++i) d4[i] = z;
    }
}

// GEMM2: col[M,576] x w_off2[576,18(pad32)] + b_off2 -> off2img[pos][18] f32
__global__ void gemm2_kernel(const _Float16* __restrict__ A, const v16h* __restrict__ Bp,
                             const float* __restrict__ bias, float* __restrict__ off2img) {
    int wave = (int)((blockIdx.x * blockDim.x + threadIdx.x) >> 5);
    int lane = threadIdx.x & 31;
    v8f acc[2];
#pragma unroll
    for (int nt = 0; nt < 2; ++nt) acc[nt] = (v8f){0,0,0,0,0,0,0,0};
    gemm_core<2, 18>(A, Bp, wave, lane, acc);
    int n0 = lane & 15;
    int mh = (lane >> 4) << 3;
#pragma unroll
    for (int nt = 0; nt < 2; ++nt) {
        int n = nt * 16 + n0;
        if (n < 18) {
            float bv = bias[n];
#pragma unroll
            for (int r = 0; r < 8; ++r) {
                int m = wave * 16 + mh + r;
                off2img[(size_t)m * 18 + n] = acc[nt][r] + bv;
            }
        }
    }
}

// ---------------------------------------------------------------------------
// Deformable im2col of h (one image) -> col[pos][576] f16
__global__ void coldef_build_kernel(const _Float16* __restrict__ h,
                                    const float* __restrict__ off2img, int b,
                                    _Float16* __restrict__ col) {
    int idx = blockIdx.x * blockDim.x + threadIdx.x;
    if (idx >= MIMG * 9) return;
    int t = idx % 9; int pos = idx / 9;
    int xw = pos % W; int yh = pos / W;
    float dy = off2img[(size_t)pos * 18 + t * 2];
    float dx = off2img[(size_t)pos * 18 + t * 2 + 1];
    float py = (float)(yh + t / 3 - 1) + dy;
    float px = (float)(xw + t % 3 - 1) + dx;
    float fy = floorf(py), fx = floorf(px);
    int y0 = (int)fy, x0 = (int)fx;
    float ly = py - fy, lx = px - fx;
    float w00 = (1.f - ly) * (1.f - lx), w01 = (1.f - ly) * lx;
    float w10 = ly * (1.f - lx),         w11 = ly * lx;
    const _Float16* hb = h + (size_t)b * MIMG * HID;
    const _Float16* g00 = (y0   >= 0 && y0   < H && x0   >= 0 && x0   < W) ? hb + ((size_t)y0       * W + x0    ) * HID : nullptr;
    const _Float16* g01 = (y0   >= 0 && y0   < H && x0+1 >= 0 && x0+1 < W) ? hb + ((size_t)y0       * W + x0 + 1) * HID : nullptr;
    const _Float16* g10 = (y0+1 >= 0 && y0+1 < H && x0   >= 0 && x0   < W) ? hb + ((size_t)(y0 + 1) * W + x0    ) * HID : nullptr;
    const _Float16* g11 = (y0+1 >= 0 && y0+1 < H && x0+1 >= 0 && x0+1 < W) ? hb + ((size_t)(y0 + 1) * W + x0 + 1) * HID : nullptr;
    _Float16* dst = col + (size_t)pos * 576 + t * HID;
    for (int c = 0; c < HID; ++c) {
        float s = 0.f;
        if (g00) s += w00 * (float)g00[c];
        if (g01) s += w01 * (float)g01[c];
        if (g10) s += w10 * (float)g10[c];
        if (g11) s += w11 * (float)g11[c];
        dst[c] = (_Float16)s;
    }
}

// GEMM3: col[M,576] x w2[576,7(pad16)] + b2 + residual -> out NHWC f32
__global__ void gemm3_kernel(const _Float16* __restrict__ A, const v16h* __restrict__ Bp,
                             const float* __restrict__ b2, const float* __restrict__ xres,
                             float* __restrict__ outimg) {
    int wave = (int)((blockIdx.x * blockDim.x + threadIdx.x) >> 5);
    int lane = threadIdx.x & 31;
    v8f acc[1];
    acc[0] = (v8f){0,0,0,0,0,0,0,0};
    gemm_core<1, 18>(A, Bp, wave, lane, acc);
    int n  = lane & 15;
    int mh = (lane >> 4) << 3;
    if (n < S) {
        float bv = b2[n];
#pragma unroll
        for (int r = 0; r < 8; ++r) {
            int m = wave * 16 + mh + r;
            outimg[(size_t)m * S + n] = acc[0][r] + bv + xres[(size_t)m * S + n];
        }
    }
}

// ---------------------------------------------------------------------------
extern "C" void kernel_launch(void* const* d_in, const int* in_sizes, int n_in,
                              void* d_out, int out_size, void* d_ws, size_t ws_size,
                              hipStream_t stream) {
    const float* x      = (const float*)d_in[0];
    const float* w_off1 = (const float*)d_in[1];
    const float* b_off1 = (const float*)d_in[2];
    const float* w1     = (const float*)d_in[3];
    const float* b1     = (const float*)d_in[4];
    const float* gamma  = (const float*)d_in[5];
    const float* beta   = (const float*)d_in[6];
    const float* mean   = (const float*)d_in[7];
    const float* var    = (const float*)d_in[8];
    const float* w_off2 = (const float*)d_in[9];
    const float* b_off2 = (const float*)d_in[10];
    const float* w2     = (const float*)d_in[11];
    const float* b2     = (const float*)d_in[12];
    float* out = (float*)d_out;

    // workspace carve (~80 MB total, fits easily in L2-sized scratch)
    size_t off = 0;
    auto carve = [&](size_t bytes) -> void* {
        void* p = (char*)d_ws + off;
        off += (bytes + 255) & ~(size_t)255;
        return p;
    };
    float*    off1b   = (float*)   carve((size_t)MPOS * 18 * 4);   // 10.6 MB
    _Float16* col1    = (_Float16*)carve((size_t)MPOS * 64 * 2);   // 18.9 MB
    _Float16* hbuf    = (_Float16*)carve((size_t)MPOS * 64 * 2);   // 18.9 MB
    float*    off2b   = (float*)   carve((size_t)MPOS * 18 * 4);   // 10.6 MB
    _Float16* colbig  = (_Float16*)carve((size_t)MIMG * 576 * 2);  // 21.2 MB (per-image reuse)
    v16h*     w1p     = (v16h*)    carve((size_t)4 * 2 * 32 * 32);
    v16h*     w2p     = (v16h*)    carve((size_t)2 * 18 * 32 * 32);
    v16h*     w3p     = (v16h*)    carve((size_t)1 * 18 * 32 * 32);
    float*    bnscale = (float*)   carve(256);
    float*    bnshift = (float*)   carve(256);

    // stage 0: constants
    bn_prep_kernel<<<1, 64, 0, stream>>>(gamma, beta, mean, var, b1, bnscale, bnshift);
    pack_weight_kernel<<<1, 256, 0, stream>>>(w1,     w1p,  7, 64, 63,  4, 2);
    pack_weight_kernel<<<5, 256, 0, stream>>>(w_off2, w2p, 64, 18, 576, 2, 18);
    pack_weight_kernel<<<3, 256, 0, stream>>>(w2,     w3p, 64,  7, 576, 1, 18);

    // stage 1: offsets 1 + deformable im2col + GEMM1 (BN+ReLU fused)
    off1_conv_kernel<<<(MPOS * 18 + 255) / 256, 256, 0, stream>>>(x, w_off1, b_off1, off1b);
    col1_build_kernel<<<(MPOS * 9 + 255) / 256, 256, 0, stream>>>(x, off1b, col1);
    gemm1_kernel<<<(MPOS / 16) * 32 / 256, 256, 0, stream>>>(col1, w1p, bnscale, bnshift, hbuf);

    // stage 2: per image (reuse colbig for std + deformed im2col)
    for (int b = 0; b < BATCH; ++b) {
        colstd_build_kernel<<<(MIMG * 9 + 255) / 256, 256, 0, stream>>>(hbuf, b, colbig);
        gemm2_kernel<<<(MIMG / 16) * 32 / 256, 256, 0, stream>>>(
            colbig, w2p, b_off2, off2b + (size_t)b * MIMG * 18);
        coldef_build_kernel<<<(MIMG * 9 + 255) / 256, 256, 0, stream>>>(
            hbuf, off2b + (size_t)b * MIMG * 18, b, colbig);
        gemm3_kernel<<<(MIMG / 16) * 32 / 256, 256, 0, stream>>>(
            colbig, w3p, b2, x + (size_t)b * MIMG * S, out + (size_t)b * MIMG * S);
    }
}
